// SwinTransformerBlock_15350213115939
// MI455X (gfx1250) — matrix operand order repaired and checked by reading, not compiled, not run on hardware
//
#include <hip/hip_runtime.h>
#include <hip/hip_bf16.h>
#include <cstdint>

// ---------------------------------------------------------------------------
// Types for CDNA5 WMMA (wave32, v_wmma_f32_16x16x32_bf16)
// ---------------------------------------------------------------------------
typedef __attribute__((ext_vector_type(16))) __bf16 v16bf;
typedef __attribute__((ext_vector_type(8)))  __bf16 bf16x8;
typedef __attribute__((ext_vector_type(8)))  float  v8f;
typedef int v4i_vs __attribute__((vector_size(16)));

union V16U { v16bf v; bf16x8 h[2]; };

__device__ __forceinline__ __bf16 f2bf(float f) {
  union { float f; uint32_t u; } a; a.f = f;
  uint32_t r = a.u + 0x7FFFu + ((a.u >> 16) & 1u);   // round-to-nearest-even
  unsigned short s = (unsigned short)(r >> 16);
  union { unsigned short s; __bf16 b; } o; o.s = s;
  return o.b;
}

// A operand (16x32, row-major, K contiguous, row stride ld elements):
// lane&15 = row M; lane>>4 selects K halves {0..7,16..23} vs {8..15,24..31}
__device__ __forceinline__ v16bf load_a(const __bf16* base, int ld, int lane) {
  int row = lane & 15;
  int kb  = (lane >> 4) * 8;
  V16U u;
  u.h[0] = *reinterpret_cast<const bf16x8*>(base + (size_t)row * ld + kb);
  u.h[1] = *reinterpret_cast<const bf16x8*>(base + (size_t)row * ld + kb + 16);
  return u.v;
}

// B operand for NT GEMM: memory holds B^T as [N][K] row-major (row stride ld).
// lane&15 = column N; lane>>4 selects K halves {0..7,16..23} vs {8..15,24..31}
__device__ __forceinline__ v16bf load_bT(const __bf16* base, int ld, int lane) {
  int col = lane & 15;
  int kb  = (lane >> 4) * 8;
  V16U u;
  u.h[0] = *reinterpret_cast<const bf16x8*>(base + (size_t)col * ld + kb);
  u.h[1] = *reinterpret_cast<const bf16x8*>(base + (size_t)col * ld + kb + 16);
  return u.v;
}

__device__ __forceinline__ v8f wmma_bf16(v16bf a, v16bf b, v8f c) {
  return __builtin_amdgcn_wmma_f32_16x16x32_bf16(false, a, false, b, (short)0, c,
                                                 false, false);
}

// --- CDNA5 async global->LDS copy (ASYNCcnt tracked), 16B per lane ----------
__device__ __forceinline__ void async_cp16(void* lds_dst, const void* gsrc) {
#if defined(__has_builtin)
#if __has_builtin(__builtin_amdgcn_global_load_async_to_lds_b128)
  __builtin_amdgcn_global_load_async_to_lds_b128(
      (__attribute__((address_space(1))) v4i_vs*)gsrc,
      (__attribute__((address_space(3))) v4i_vs*)lds_dst, 0, 0);
  return;
#else
  *(float4*)lds_dst = *(const float4*)gsrc;      // sync fallback
#endif
#else
  *(float4*)lds_dst = *(const float4*)gsrc;
#endif
}

__device__ __forceinline__ void wait_async0() {
#if defined(__has_builtin)
#if __has_builtin(__builtin_amdgcn_s_wait_asynccnt)
  __builtin_amdgcn_s_wait_asynccnt(0);
  return;
#else
  asm volatile("s_wait_asynccnt 0" ::: "memory");
#endif
#else
  asm volatile("s_wait_asynccnt 0" ::: "memory");
#endif
}

__device__ __forceinline__ float red16_max(float v) {
  v = fmaxf(v, __shfl_xor(v, 1, 32));
  v = fmaxf(v, __shfl_xor(v, 2, 32));
  v = fmaxf(v, __shfl_xor(v, 4, 32));
  v = fmaxf(v, __shfl_xor(v, 8, 32));
  return v;
}
__device__ __forceinline__ float red16_sum(float v) {
  v += __shfl_xor(v, 1, 32);
  v += __shfl_xor(v, 2, 32);
  v += __shfl_xor(v, 4, 32);
  v += __shfl_xor(v, 8, 32);
  return v;
}

// ---------------------------------------------------------------------------
// Problem constants
// ---------------------------------------------------------------------------
#define NTOK   131072      // B*S*H*W
#define CCH    192
#define NWIN   256
#define NWTOK  512
#define NHEAD  6
#define HDIM   32

// ---------------------------------------------------------------------------
// fp32 -> bf16 weight conversion
// ---------------------------------------------------------------------------
__global__ void cvt_kernel(const float* __restrict__ s, __bf16* __restrict__ d, int n) {
  int i = blockIdx.x * 256 + threadIdx.x;
  if (i < n) d[i] = f2bf(s[i]);
}

// ---------------------------------------------------------------------------
// LN1 + window partition + pos_emb  -> xw bf16 [nW,512,192]; one wave/token
// ---------------------------------------------------------------------------
__global__ void ln1_win_kernel(const float* __restrict__ x, const float* __restrict__ g,
                               const float* __restrict__ bta, const float* __restrict__ pos,
                               __bf16* __restrict__ xw) {
  int lane = threadIdx.x & 31;
  int wv   = threadIdx.x >> 5;
  size_t t = (size_t)blockIdx.x * 8 + wv;
  const float* xp = x + t * CCH;
  float v[6];
  float s = 0.f, ss = 0.f;
#pragma unroll
  for (int i = 0; i < 6; ++i) { v[i] = xp[lane + i * 32]; s += v[i]; ss += v[i] * v[i]; }
#pragma unroll
  for (int m = 1; m < 32; m <<= 1) { s += __shfl_xor(s, m, 32); ss += __shfl_xor(ss, m, 32); }
  float mu   = s  * (1.f / 192.f);
  float var  = ss * (1.f / 192.f) - mu * mu;
  float rstd = rsqrtf(var + 1e-5f);

  int w_ = (int)(t & 63), h = (int)((t >> 6) & 63), sp = (int)((t >> 12) & 15), b = (int)(t >> 16);
  int wi = ((b * 2 + (sp >> 3)) * 8 + (h >> 3)) * 8 + (w_ >> 3);
  int n  = (((sp & 7) * 8) + (h & 7)) * 8 + (w_ & 7);
  __bf16* op = xw + ((size_t)wi * NWTOK + n) * CCH;
  const float* pp = pos + (size_t)n * CCH;
#pragma unroll
  for (int i = 0; i < 6; ++i) {
    int c = lane + i * 32;
    op[c] = f2bf((v[i] - mu) * rstd * g[c] + bta[c] + pp[c]);
  }
}

// ---------------------------------------------------------------------------
// LN2: xres fp32 -> hbuf bf16 [tok,192]
// ---------------------------------------------------------------------------
__global__ void ln2_kernel(const float* __restrict__ xr, const float* __restrict__ g,
                           const float* __restrict__ bta, __bf16* __restrict__ hb) {
  int lane = threadIdx.x & 31;
  int wv   = threadIdx.x >> 5;
  size_t t = (size_t)blockIdx.x * 8 + wv;
  const float* xp = xr + t * CCH;
  float v[6];
  float s = 0.f, ss = 0.f;
#pragma unroll
  for (int i = 0; i < 6; ++i) { v[i] = xp[lane + i * 32]; s += v[i]; ss += v[i] * v[i]; }
#pragma unroll
  for (int m = 1; m < 32; m <<= 1) { s += __shfl_xor(s, m, 32); ss += __shfl_xor(ss, m, 32); }
  float mu   = s  * (1.f / 192.f);
  float var  = ss * (1.f / 192.f) - mu * mu;
  float rstd = rsqrtf(var + 1e-5f);
  __bf16* op = hb + t * CCH;
#pragma unroll
  for (int i = 0; i < 6; ++i) {
    int c = lane + i * 32;
    op[c] = f2bf((v[i] - mu) * rstd * g[c] + bta[c]);
  }
}

// ---------------------------------------------------------------------------
// QKV GEMM with async-staged weight tile in LDS.
// Block = 8 waves sharing (w, nt): stage wq[nt*64 .. nt*64+63][0..191] (24KB),
// wave wv handles mtile = mg*8+wv (16x64 tile).
// grid = 256 * 4 * 9 = 9216 blocks, 256 threads.
// Q/K -> [w][head][tok][hd], V -> [w][head][hd][tok] (transposed), all bf16.
// ---------------------------------------------------------------------------
__global__ void qkv_kernel(const __bf16* __restrict__ xw, const __bf16* __restrict__ wq,
                           const float* __restrict__ qb_bias,
                           __bf16* __restrict__ qb, __bf16* __restrict__ kb,
                           __bf16* __restrict__ vT) {
  __shared__ alignas(16) __bf16 wt[64 * CCH];    // 24KB weight tile
  int lane = threadIdx.x & 31;
  int wv   = threadIdx.x >> 5;
  int bid  = blockIdx.x;
  int nt   = bid % 9;
  int rem  = bid / 9;
  int mg   = rem & 3;
  int w    = rem >> 2;
  int mt   = mg * 8 + wv;

  // cooperative async stage: 64*192 bf16 = 1536 x 16B chunks, 6 per thread
  {
    const char* src = (const char*)(wq + (size_t)(nt * 64) * CCH);
    char*       dst = (char*)wt;
#pragma unroll
    for (int i = 0; i < 6; ++i) {
      int idx = threadIdx.x + i * 256;
      async_cp16(dst + idx * 16, src + idx * 16);
    }
    wait_async0();
    __syncthreads();
  }

  v8f acc[4];
#pragma unroll
  for (int j = 0; j < 4; ++j) {
    float bias = qb_bias[nt * 64 + j * 16 + (lane & 15)];
#pragma unroll
    for (int r = 0; r < 8; ++r) acc[j][r] = bias;
  }
  const __bf16* ap = xw + ((size_t)w * NWTOK + mt * 16) * CCH;
#pragma unroll
  for (int kc = 0; kc < 6; ++kc) {
    v16bf a = load_a(ap + kc * 32, CCH, lane);
#pragma unroll
    for (int j = 0; j < 4; ++j) {
      v16bf b = load_bT(wt + (size_t)(j * 16) * CCH + kc * 32, CCH, lane);
      acc[j] = wmma_bf16(a, b, acc[j]);
    }
  }
#pragma unroll
  for (int j = 0; j < 4; ++j) {
    int n     = nt * 64 + j * 16 + (lane & 15);
    int which = n / CCH;
    int hm    = n - which * CCH;
    int head  = hm >> 5;
    int hd    = hm & 31;
#pragma unroll
    for (int r = 0; r < 8; ++r) {
      int token = mt * 16 + r + 8 * (lane >> 4);
      __bf16 val = f2bf(acc[j][r]);
      if (which == 0)      qb[((size_t)(w * NHEAD + head) * NWTOK + token) * HDIM + hd] = val;
      else if (which == 1) kb[((size_t)(w * NHEAD + head) * NWTOK + token) * HDIM + hd] = val;
      else                 vT[((size_t)(w * NHEAD + head) * HDIM + hd) * NWTOK + token] = val;
    }
  }
}

// ---------------------------------------------------------------------------
// Flash attention with async-staged K panel.
// Block = 8 waves sharing (w, h); stage whole K [512,32] bf16 (32KB) via
// async-to-LDS; wave wv handles q rows (mg*8+wv)*16..+15. V read from global.
// grid = 256*6*4 = 6144 blocks, 256 threads. LDS = 32KB K + 8KB P = 40KB.
// ---------------------------------------------------------------------------
__global__ void attn_kernel(const __bf16* __restrict__ qb, const __bf16* __restrict__ kb,
                            const __bf16* __restrict__ vT, __bf16* __restrict__ obuf) {
  __shared__ alignas(16) __bf16 kt[NWTOK * HDIM];   // 32KB
  __shared__ alignas(16) __bf16 pst[8][16 * 32];    // 8KB
  int lane = threadIdx.x & 31;
  int wv   = threadIdx.x >> 5;
  int bid  = blockIdx.x;
  int mg   = bid & 3;
  int wh   = bid >> 2;
  int h    = wh % NHEAD;
  int w    = wh / NHEAD;
  int mt   = mg * 8 + wv;

  const __bf16* qp = qb + (size_t)(w * NHEAD + h) * NWTOK * HDIM;
  const __bf16* kp = kb + (size_t)(w * NHEAD + h) * NWTOK * HDIM;
  const __bf16* vp = vT + (size_t)(w * NHEAD + h) * HDIM * NWTOK;

  // cooperative async stage of K: 512*32 bf16 = 2048 x 16B chunks, 8/thread
  {
    const char* src = (const char*)kp;
    char*       dst = (char*)kt;
#pragma unroll
    for (int i = 0; i < 8; ++i) {
      int idx = threadIdx.x + i * 256;
      async_cp16(dst + idx * 16, src + idx * 16);
    }
    wait_async0();
    __syncthreads();
  }

  v16bf aq = load_a(qp + (size_t)mt * 16 * HDIM, HDIM, lane);

  float mrow[8], lrow[8];
  v8f o0, o1, zro;
#pragma unroll
  for (int r = 0; r < 8; ++r) { mrow[r] = -1e30f; lrow[r] = 0.f; o0[r] = 0.f; o1[r] = 0.f; zro[r] = 0.f; }
  const float scale = 0.17677669529663687f;   // 32^-0.5
  __bf16* ps = pst[wv];

  for (int ch = 0; ch < 16; ++ch) {
    int tok = ch * 32;
    v16bf b0 = load_bT(kt + (size_t)tok * HDIM, HDIM, lane);
    v16bf b1 = load_bT(kt + (size_t)(tok + 16) * HDIM, HDIM, lane);
    v8f s0 = wmma_bf16(aq, b0, zro);
    v8f s1 = wmma_bf16(aq, b1, zro);
#pragma unroll
    for (int r = 0; r < 8; ++r) {
      float x0 = s0[r] * scale;
      float x1 = s1[r] * scale;
      float tm = red16_max(fmaxf(x0, x1));
      float nm = fmaxf(mrow[r], tm);
      float al = __expf(mrow[r] - nm);
      float p0 = __expf(x0 - nm);
      float p1 = __expf(x1 - nm);
      lrow[r] = lrow[r] * al + red16_sum(p0 + p1);
      mrow[r] = nm;
      o0[r] *= al;
      o1[r] *= al;
      int m = r + 8 * (lane >> 4);
      int n = lane & 15;
      ps[m * 32 + n]      = f2bf(p0);
      ps[m * 32 + 16 + n] = f2bf(p1);
    }
    asm volatile("s_wait_dscnt 0" ::: "memory");
    v16bf ap  = load_a(ps, 32, lane);
    v16bf bv0 = load_bT(vp + tok, NWTOK, lane);                      // hd 0..15
    v16bf bv1 = load_bT(vp + (size_t)16 * NWTOK + tok, NWTOK, lane); // hd 16..31
    o0 = wmma_bf16(ap, bv0, o0);
    o1 = wmma_bf16(ap, bv1, o1);
  }
#pragma unroll
  for (int r = 0; r < 8; ++r) {
    float inv = 1.0f / lrow[r];
    int token = mt * 16 + r + 8 * (lane >> 4);
    int c0    = h * HDIM + (lane & 15);
    __bf16* op = obuf + ((size_t)w * NWTOK + token) * CCH;
    op[c0]      = f2bf(o0[r] * inv);
    op[c0 + 16] = f2bf(o1[r] * inv);
  }
}

// ---------------------------------------------------------------------------
// proj GEMM + window reverse + residual -> xres fp32 [tok,192]
// Block = 8 waves sharing (w, nt); async-staged 24KB weight tile.
// grid = 256*4*3 = 3072 blocks, 256 threads.
// ---------------------------------------------------------------------------
__global__ void proj_kernel(const __bf16* __restrict__ obuf, const __bf16* __restrict__ pw,
                            const float* __restrict__ pb, const float* __restrict__ x,
                            float* __restrict__ xres) {
  __shared__ alignas(16) __bf16 wt[64 * CCH];    // 24KB
  int lane = threadIdx.x & 31;
  int wv   = threadIdx.x >> 5;
  int bid  = blockIdx.x;
  int nt   = bid % 3;
  int rem  = bid / 3;
  int mg   = rem & 3;
  int w    = rem >> 2;
  int mt   = mg * 8 + wv;

  {
    const char* src = (const char*)(pw + (size_t)(nt * 64) * CCH);
    char*       dst = (char*)wt;
#pragma unroll
    for (int i = 0; i < 6; ++i) {
      int idx = threadIdx.x + i * 256;
      async_cp16(dst + idx * 16, src + idx * 16);
    }
    wait_async0();
    __syncthreads();
  }

  v8f acc[4];
#pragma unroll
  for (int j = 0; j < 4; ++j) {
    float bias = pb[nt * 64 + j * 16 + (lane & 15)];
#pragma unroll
    for (int r = 0; r < 8; ++r) acc[j][r] = bias;
  }
  const __bf16* ap = obuf + ((size_t)w * NWTOK + mt * 16) * CCH;
#pragma unroll
  for (int kc = 0; kc < 6; ++kc) {
    v16bf a = load_a(ap + kc * 32, CCH, lane);
#pragma unroll
    for (int j = 0; j < 4; ++j) {
      v16bf b = load_bT(wt + (size_t)(j * 16) * CCH + kc * 32, CCH, lane);
      acc[j] = wmma_bf16(a, b, acc[j]);
    }
  }
  int b_ = w >> 7, sg = (w >> 6) & 1, hg = (w >> 3) & 7, wg = w & 7;
#pragma unroll
  for (int r = 0; r < 8; ++r) {
    int n  = mt * 16 + r + 8 * (lane >> 4);
    int ls = n >> 6, lh = (n >> 3) & 7, lw = n & 7;
    size_t t = (((size_t)b_ * 16 + sg * 8 + ls) * 64 + hg * 8 + lh) * 64 + wg * 8 + lw;
#pragma unroll
    for (int j = 0; j < 4; ++j) {
      int c = nt * 64 + j * 16 + (lane & 15);
      xres[t * CCH + c] = x[t * CCH + c] + acc[j][r];
    }
  }
}

// ---------------------------------------------------------------------------
// Fused MLP: fc1 + GELU + fc2 + residual. Wave = 16 tokens; hidden 16x768
// tile lives in LDS (24KB/wave). 2 waves/block, tasks = 8192.
// ---------------------------------------------------------------------------
__global__ void mlp_kernel(const float* __restrict__ xres, const __bf16* __restrict__ hbuf,
                           const __bf16* __restrict__ w1, const float* __restrict__ b1,
                           const __bf16* __restrict__ w2, const float* __restrict__ b2,
                           float* __restrict__ out) {
  __shared__ alignas(16) __bf16 mid[2][16 * 768];
  int lane  = threadIdx.x & 31;
  int wv    = threadIdx.x >> 5;
  int task  = blockIdx.x * 2 + wv;
  int mbase = task * 16;
  __bf16* ms = mid[wv];

  v16bf a6[6];
#pragma unroll
  for (int kc = 0; kc < 6; ++kc)
    a6[kc] = load_a(hbuf + (size_t)mbase * CCH + kc * 32, CCH, lane);

  // fc1 + GELU -> LDS
  for (int nt = 0; nt < 48; ++nt) {
    float bias = b1[nt * 16 + (lane & 15)];
    v8f acc;
#pragma unroll
    for (int r = 0; r < 8; ++r) acc[r] = bias;
#pragma unroll
    for (int kc = 0; kc < 6; ++kc) {
      v16bf b = load_bT(w1 + (size_t)(nt * 16) * CCH + kc * 32, CCH, lane);
      acc = wmma_bf16(a6[kc], b, acc);
    }
#pragma unroll
    for (int r = 0; r < 8; ++r) {
      float v = acc[r];
      float ge = 0.5f * v * (1.0f + erff(v * 0.7071067811865475f));
      ms[(r + 8 * (lane >> 4)) * 768 + nt * 16 + (lane & 15)] = f2bf(ge);
    }
  }
  asm volatile("s_wait_dscnt 0" ::: "memory");

  // fc2 + residual -> out
  for (int nt = 0; nt < 12; ++nt) {
    float bias = b2[nt * 16 + (lane & 15)];
    v8f acc;
#pragma unroll
    for (int r = 0; r < 8; ++r) acc[r] = bias;
    for (int kc = 0; kc < 24; ++kc) {
      v16bf a = load_a(ms + kc * 32, 768, lane);
      v16bf b = load_bT(w2 + (size_t)(nt * 16) * 768 + kc * 32, 768, lane);
      acc = wmma_bf16(a, b, acc);
    }
#pragma unroll
    for (int r = 0; r < 8; ++r) {
      size_t t = (size_t)mbase + r + 8 * (lane >> 4);
      int c = nt * 16 + (lane & 15);
      out[t * CCH + c] = xres[t * CCH + c] + acc[r];
    }
  }
}

// ---------------------------------------------------------------------------
// Host launcher
// ---------------------------------------------------------------------------
extern "C" void kernel_launch(void* const* d_in, const int* in_sizes, int n_in,
                              void* d_out, int out_size, void* d_ws, size_t ws_size,
                              hipStream_t stream) {
  const float* x      = (const float*)d_in[0];
  const float* n1g    = (const float*)d_in[1];
  const float* n1b    = (const float*)d_in[2];
  const float* pos    = (const float*)d_in[3];
  const float* qkvw_f = (const float*)d_in[4];
  const float* qkvb   = (const float*)d_in[5];
  const float* projw_f= (const float*)d_in[6];
  const float* projb  = (const float*)d_in[7];
  const float* n2g    = (const float*)d_in[8];
  const float* n2b    = (const float*)d_in[9];
  const float* fc1w_f = (const float*)d_in[10];
  const float* fc1b   = (const float*)d_in[11];
  const float* fc2w_f = (const float*)d_in[12];
  const float* fc2b   = (const float*)d_in[13];
  float* out = (float*)d_out;

  // workspace layout (bytes)
  const size_t SZ_XW   = (size_t)NWIN * NWTOK * CCH * 2;          // xw / obuf
  const size_t SZ_QKV1 = (size_t)NWIN * NHEAD * NWTOK * HDIM * 2; // q / k / vT
  const size_t SZ_XRES = (size_t)NTOK * CCH * 4;
  const size_t SZ_H    = (size_t)NTOK * CCH * 2;
  char* ws = (char*)d_ws;
  __bf16* xw    = (__bf16*)(ws);                                  // also obuf
  __bf16* qb    = (__bf16*)(ws + SZ_XW);
  __bf16* kb    = (__bf16*)(ws + SZ_XW + SZ_QKV1);
  __bf16* vT    = (__bf16*)(ws + SZ_XW + 2 * SZ_QKV1);
  float*  xres  = (float*) (ws + SZ_XW + 3 * SZ_QKV1);
  __bf16* hbuf  = (__bf16*)(ws + SZ_XW + 3 * SZ_QKV1 + SZ_XRES);
  char*   wbase = ws + SZ_XW + 3 * SZ_QKV1 + SZ_XRES + SZ_H;
  __bf16* wqkv  = (__bf16*)(wbase);
  __bf16* wproj = (__bf16*)(wbase + 576 * 192 * 2);
  __bf16* wfc1  = (__bf16*)(wbase + 576 * 192 * 2 + 192 * 192 * 2);
  __bf16* wfc2  = (__bf16*)(wbase + 576 * 192 * 2 + 192 * 192 * 2 + 768 * 192 * 2);

  // weight conversion
  cvt_kernel<<<(576 * 192 + 255) / 256, 256, 0, stream>>>(qkvw_f, wqkv, 576 * 192);
  cvt_kernel<<<(192 * 192 + 255) / 256, 256, 0, stream>>>(projw_f, wproj, 192 * 192);
  cvt_kernel<<<(768 * 192 + 255) / 256, 256, 0, stream>>>(fc1w_f, wfc1, 768 * 192);
  cvt_kernel<<<(192 * 768 + 255) / 256, 256, 0, stream>>>(fc2w_f, wfc2, 192 * 768);

  // pipeline
  ln1_win_kernel<<<NTOK / 8, 256, 0, stream>>>(x, n1g, n1b, pos, xw);
  qkv_kernel<<<NWIN * 4 * 9, 256, 0, stream>>>(xw, wqkv, qkvb, qb, kb, vT);
  attn_kernel<<<NWIN * NHEAD * 4, 256, 0, stream>>>(qb, kb, vT, xw /*obuf*/);
  proj_kernel<<<NWIN * 4 * 3, 256, 0, stream>>>(xw /*obuf*/, wproj, projb, x, xres);
  ln2_kernel<<<NTOK / 8, 256, 0, stream>>>(xres, n2g, n2b, hbuf);
  mlp_kernel<<<(NTOK / 16) / 2, 64, 0, stream>>>(xres, hbuf, wfc1, fc1b, wfc2, fc2b, out);
}
